// FlashSVDBlock_85529978733253
// MI455X (gfx1250) — compile-verified, matrix operand order checked by reading, not run on hardware
//
#include <hip/hip_runtime.h>
#include <hip/hip_bf16.h>
#include <stdint.h>

// ---------------------------------------------------------------------------
// FlashSVD transformer block for MI455X (gfx1250), bf16 WMMA pipeline.
// Round 3: fix async global->LDS builtin argument types (int4 AS1 / AS3).
// ---------------------------------------------------------------------------

typedef __bf16 bf16;
typedef __attribute__((ext_vector_type(16))) __bf16 v16bf;
typedef __attribute__((ext_vector_type(8)))  __bf16 v8bf;
typedef __attribute__((ext_vector_type(8)))  float  v8f;

#define C_B    4
#define C_M    2048
#define C_D    768
#define C_H    12
#define C_DH   64
#define C_DFF  3072
#define C_R    32
#define C_BM   8192   // B*M

// ---------------------------------------------------------------------------
// gfx1250 async global->LDS copy (guarded so either toolchain compiles)
// ---------------------------------------------------------------------------
#if defined(__has_builtin)
# if __has_builtin(__builtin_amdgcn_global_load_async_to_lds_b128)
#  define HAVE_ASYNC_COPY 1
# endif
#endif
#ifndef HAVE_ASYNC_COPY
# define HAVE_ASYNC_COPY 0
#endif

#if HAVE_ASYNC_COPY
typedef int v4i_gcc __attribute__((vector_size(16)));
typedef __attribute__((address_space(1))) v4i_gcc as1_v4i;
typedef __attribute__((address_space(3))) v4i_gcc as3_v4i;
# if __has_builtin(__builtin_amdgcn_s_wait_asynccnt)
#  define WAIT_ASYNC(n) __builtin_amdgcn_s_wait_asynccnt(n)
# else
#  define WAIT_ASYNC(n) asm volatile("s_wait_asynccnt %0" :: "n"(n) : "memory")
# endif
#else
# define WAIT_ASYNC(n) ((void)0)
#endif

// Copy 16 bytes global -> LDS (async if available, else sync load + ds_store).
__device__ __forceinline__ void stage16(const bf16* g, bf16* l) {
#if HAVE_ASYNC_COPY
  __builtin_amdgcn_global_load_async_to_lds_b128((as1_v4i*)g, (as3_v4i*)l, 0, 0);
#else
  *(v8bf*)l = *(const v8bf*)g;
#endif
}

// ---------------------------------------------------------------------------
// WMMA helpers
// ---------------------------------------------------------------------------
__device__ __forceinline__ v8f wmma_bf16(v16bf a, v16bf b, v8f c) {
  return __builtin_amdgcn_wmma_f32_16x16x32_bf16(
      false, a, false, b, (short)0, c, false, false);
}

// 16x32 bf16 fragment from row-major [rows][ld] (A layout; also B when the
// matrix is stored [N][K]). Lane L: row = row0+(L&15), half = L>>4;
// elems 0..7 <- k0+8*half.., elems 8..15 <- k0+16+8*half.. (2x16B loads).
__device__ __forceinline__ v16bf load_frag(const bf16* __restrict__ base,
                                           int ld, int row0, int k0) {
  int lane = threadIdx.x & 31;
  const bf16* p = base + (size_t)(row0 + (lane & 15)) * ld + k0 + ((lane >> 4) << 3);
  v8bf lo = *(const v8bf*)(p);
  v8bf hi = *(const v8bf*)(p + 16);
  return __builtin_shufflevector(lo, hi, 0,1,2,3,4,5,6,7,8,9,10,11,12,13,14,15);
}

// Same pattern from an LDS tile stored [16][32] bf16 (row stride 32).
__device__ __forceinline__ v16bf load_frag_lds(const bf16* base) {
  int lane = threadIdx.x & 31;
  const bf16* p = base + (lane & 15) * 32 + ((lane >> 4) << 3);
  v8bf lo = *(const v8bf*)(p);
  v8bf hi = *(const v8bf*)(p + 16);
  return __builtin_shufflevector(lo, hi, 0,1,2,3,4,5,6,7,8,9,10,11,12,13,14,15);
}

__device__ __forceinline__ float wave16_max(float v) {
  #pragma unroll
  for (int m = 1; m < 16; m <<= 1) v = fmaxf(v, __shfl_xor(v, m, 32));
  return v;
}
__device__ __forceinline__ float wave16_sum(float v) {
  #pragma unroll
  for (int m = 1; m < 16; m <<= 1) v += __shfl_xor(v, m, 32);
  return v;
}

// ---------------------------------------------------------------------------
// Prep kernels
// ---------------------------------------------------------------------------
__global__ void cvt_bf16_kernel(const float* __restrict__ s, bf16* __restrict__ d, size_t n) {
  size_t i = (size_t)blockIdx.x * 256 + threadIdx.x;
  if (i < n) d[i] = (bf16)s[i];
}

__global__ void transpose_bf16_kernel(const float* __restrict__ src, bf16* __restrict__ dst,
                                      int R, int C) {
  size_t i = (size_t)blockIdx.x * 256 + threadIdx.x;
  if (i >= (size_t)R * C) return;
  int r = (int)(i / C), c = (int)(i % C);
  dst[(size_t)c * R + r] = (bf16)src[i];
}

// Wt[t][h][e][d] = sum_r P_t[h][d][r] * V_t[h][r][e]
__global__ void prep_wqkv_kernel(const float* __restrict__ Pq, const float* __restrict__ Vq,
                                 const float* __restrict__ Pk, const float* __restrict__ Vk,
                                 const float* __restrict__ Pv, const float* __restrict__ Vv,
                                 bf16* __restrict__ Wt) {
  size_t i = (size_t)blockIdx.x * 256 + threadIdx.x;
  if (i >= (size_t)3 * C_H * C_DH * C_D) return;
  int d = (int)(i % C_D);
  int e = (int)((i / C_D) % C_DH);
  int h = (int)((i / ((size_t)C_D * C_DH)) % C_H);
  int t = (int)(i / ((size_t)C_D * C_DH * C_H));
  const float* P = (t == 0) ? Pq : (t == 1) ? Pk : Pv;
  const float* V = (t == 0) ? Vq : (t == 1) ? Vk : Vv;
  float s = 0.f;
  #pragma unroll 8
  for (int r = 0; r < C_R; ++r)
    s += P[((size_t)h * C_D + d) * C_R + r] * V[((size_t)h * C_R + r) * C_DH + e];
  Wt[i] = (bf16)s;
}

// ---------------------------------------------------------------------------
// QKV projection (register double-buffered). grid=(BM/128, 3H), block=256.
// Q,K written natural [bh][m][e]; V transposed [bh][e][m].
// ---------------------------------------------------------------------------
__global__ void qkv_kernel(const bf16* __restrict__ xb, const bf16* __restrict__ Wt,
                           const float* __restrict__ bq, const float* __restrict__ bk,
                           const float* __restrict__ bv,
                           bf16* __restrict__ Qb, bf16* __restrict__ Kb, bf16* __restrict__ Vtb) {
  int w  = threadIdx.x >> 5;
  int ln = threadIdx.x & 15;
  int hf = (threadIdx.x & 31) >> 4;
  int t  = blockIdx.y / C_H;
  int h  = blockIdx.y % C_H;
  int m0 = blockIdx.x * 128 + w * 16;
  const bf16* Bt = Wt + ((size_t)(t * C_H + h) * C_DH) * C_D;  // [DH][D]
  const float* bias = ((t == 0) ? bq : (t == 1) ? bk : bv) + h * C_DH;

  v8f z = {0.f,0.f,0.f,0.f,0.f,0.f,0.f,0.f};
  v8f acc[4] = {z, z, z, z};
  v16bf a = load_frag(xb, C_D, m0, 0);
  v16bf bf[4];
  #pragma unroll
  for (int u = 0; u < 4; ++u) bf[u] = load_frag(Bt, C_D, 16 * u, 0);

  for (int k0 = 0; k0 < C_D; k0 += 32) {
    v16bf an = a, bn[4] = {bf[0], bf[1], bf[2], bf[3]};
    if (k0 + 32 < C_D) {
      an = load_frag(xb, C_D, m0, k0 + 32);
      #pragma unroll
      for (int u = 0; u < 4; ++u) bn[u] = load_frag(Bt, C_D, 16 * u, k0 + 32);
    }
    #pragma unroll
    for (int u = 0; u < 4; ++u) acc[u] = wmma_bf16(a, bf[u], acc[u]);
    a = an;
    #pragma unroll
    for (int u = 0; u < 4; ++u) bf[u] = bn[u];
  }
  #pragma unroll
  for (int u = 0; u < 4; ++u) {
    int e = 16 * u + ln;
    float be = bias[e];
    #pragma unroll
    for (int r = 0; r < 8; ++r) {
      int R = m0 + r + 8 * hf;
      int b = R >> 11;
      int m = R & (C_M - 1);
      float v = acc[u][r] + be;
      size_t bh = (size_t)b * C_H + h;
      if (t == 0)      Qb [(bh * C_M + m) * C_DH + e] = (bf16)v;
      else if (t == 1) Kb [(bh * C_M + m) * C_DH + e] = (bf16)v;
      else             Vtb[(bh * C_DH + e) * C_M + m] = (bf16)v;
    }
  }
}

// ---------------------------------------------------------------------------
// Flash attention, software-pipelined: next-step K fragments and current-step
// V fragments are issued before the softmax VALU work so their latency hides.
// grid=(M/128, B*H), block=256.
// ---------------------------------------------------------------------------
__global__ void attn_kernel(const bf16* __restrict__ Q, const bf16* __restrict__ K,
                            const bf16* __restrict__ Vt, const uint8_t* __restrict__ mask,
                            bf16* __restrict__ attnb) {
  __shared__ bf16 pbuf[8][16][40];   // per-wave 16x32 P tile (padded rows)
  int w  = threadIdx.x >> 5;
  int ln = threadIdx.x & 15;
  int hf = (threadIdx.x & 31) >> 4;
  int bh = blockIdx.y;
  int b  = bh / C_H, h = bh % C_H;
  int m0 = blockIdx.x * 128 + w * 16;

  const bf16* Qp  = Q  + (size_t)bh * C_M * C_DH;
  const bf16* Kp  = K  + (size_t)bh * C_M * C_DH;
  const bf16* Vtp = Vt + (size_t)bh * C_DH * C_M;
  const uint8_t* mp = mask + (size_t)b * C_M;

  v16bf qa0 = load_frag(Qp, C_DH, m0, 0);
  v16bf qa1 = load_frag(Qp, C_DH, m0, 32);

  v8f z = {0.f,0.f,0.f,0.f,0.f,0.f,0.f,0.f};
  v8f acc[4] = {z, z, z, z};
  float mr[8], lr[8];
  #pragma unroll
  for (int r = 0; r < 8; ++r) { mr[r] = -3.0e38f; lr[r] = 0.f; }

  v16bf kc[4];
  kc[0] = load_frag(Kp, C_DH, 0,  0);   kc[1] = load_frag(Kp, C_DH, 0,  32);
  kc[2] = load_frag(Kp, C_DH, 16, 0);   kc[3] = load_frag(Kp, C_DH, 16, 32);

  for (int n0 = 0; n0 < C_M; n0 += 32) {
    // ---- issue current V fragments + next K fragments (overlap softmax)
    v16bf vf[4];
    #pragma unroll
    for (int u = 0; u < 4; ++u) vf[u] = load_frag(Vtp, C_M, 16 * u, n0);
    v16bf kn[4] = {kc[0], kc[1], kc[2], kc[3]};
    if (n0 + 32 < C_M) {
      kn[0] = load_frag(Kp, C_DH, n0 + 32, 0);  kn[1] = load_frag(Kp, C_DH, n0 + 32, 32);
      kn[2] = load_frag(Kp, C_DH, n0 + 48, 0);  kn[3] = load_frag(Kp, C_DH, n0 + 48, 32);
    }

    // ---- scores: 16x32 tile, 4 WMMA
    v8f s0 = z, s1 = z;
    s0 = wmma_bf16(qa0, kc[0], s0);  s0 = wmma_bf16(qa1, kc[1], s0);
    s1 = wmma_bf16(qa0, kc[2], s1);  s1 = wmma_bf16(qa1, kc[3], s1);

    float madd0 = mp[n0 + ln]      ? 0.f : -1.0e30f;
    float madd1 = mp[n0 + 16 + ln] ? 0.f : -1.0e30f;

    // ---- online softmax (row r+8*hf lives in VGPR r of this half)
    #pragma unroll
    for (int r = 0; r < 8; ++r) {
      float a0 = s0[r] * 0.125f + madd0;
      float a1 = s1[r] * 0.125f + madd1;
      float rowmax = wave16_max(fmaxf(a0, a1));
      float mnew = fmaxf(mr[r], rowmax);
      float corr = __expf(mr[r] - mnew);
      float p0 = __expf(a0 - mnew);
      float p1 = __expf(a1 - mnew);
      float rs = wave16_sum(p0 + p1);
      lr[r] = lr[r] * corr + rs;
      mr[r] = mnew;
      #pragma unroll
      for (int u = 0; u < 4; ++u) acc[u][r] *= corr;
      int mrow = r + 8 * hf;
      pbuf[w][mrow][ln]      = (bf16)p0;
      pbuf[w][mrow][16 + ln] = (bf16)p1;
    }
    __syncthreads();

    // ---- transpose P via LDS, then P(16x32) @ V(32x64): 4 WMMA
    v16bf pa;
    {
      const bf16* lp = &pbuf[w][ln][hf << 3];
      v8bf lo = *(const v8bf*)(lp);
      v8bf hi = *(const v8bf*)(lp + 16);
      pa = __builtin_shufflevector(lo, hi, 0,1,2,3,4,5,6,7,8,9,10,11,12,13,14,15);
    }
    #pragma unroll
    for (int u = 0; u < 4; ++u) acc[u] = wmma_bf16(pa, vf[u], acc[u]);
    __syncthreads();

    #pragma unroll
    for (int u = 0; u < 4; ++u) kc[u] = kn[u];
  }

  #pragma unroll
  for (int r = 0; r < 8; ++r) {
    float inv = 1.0f / lr[r];
    int mloc = m0 + r + 8 * hf;
    size_t rowbase = ((size_t)b * C_M + mloc) * C_D + h * C_DH;
    #pragma unroll
    for (int u = 0; u < 4; ++u)
      attnb[rowbase + 16 * u + ln] = (bf16)(acc[u][r] * inv);
  }
}

// ---------------------------------------------------------------------------
// Generic bf16 WMMA GEMM with LDS-staged, double-buffered B tiles.
// A[BM x K] * B[K x N] (B stored transposed [N][K]).
// Each K-step: 256 threads async-copy the 64x32 B tile (4KB) to LDS while the
// previous tile feeds 4 WMMA per wave; A fragments are register-prefetched.
// EPI: 0 = bf16 out, 2 = +bias +exact-GELU bf16 out, 3 = +bias +residual f32.
// grid = (BM/128, N/64), block = 256.
// ---------------------------------------------------------------------------
template <int EPI>
__global__ void gemm_kernel(const bf16* __restrict__ A, const bf16* __restrict__ Bt,
                            const float* __restrict__ bias, const float* __restrict__ resid,
                            float* __restrict__ outf, bf16* __restrict__ outb,
                            int N, int Kdim) {
  __shared__ bf16 bsh[2][64][32];    // double-buffered B tile
  int tid = threadIdx.x;
  int w  = tid >> 5;
  int ln = tid & 15;
  int hf = (tid & 31) >> 4;
  int m0 = blockIdx.x * 128 + w * 16;
  int n0 = blockIdx.y * 64;

  // Staging assignment: thread tid copies 16B: B row n0+(tid>>2), k-seg tid&3.
  const bf16* bsrc = Bt + (size_t)(n0 + (tid >> 2)) * Kdim + ((tid & 3) << 3);
  bf16* ldst[2] = { &bsh[0][tid >> 2][(tid & 3) << 3],
                    &bsh[1][tid >> 2][(tid & 3) << 3] };

  v8f z = {0.f,0.f,0.f,0.f,0.f,0.f,0.f,0.f};
  v8f acc[4] = {z, z, z, z};

  stage16(bsrc, ldst[0]);                       // prologue: tile k0=0
  v16bf a = load_frag(A, Kdim, m0, 0);

  int buf = 0;
  for (int k0 = 0; k0 < Kdim; k0 += 32) {
    v16bf an = a;
    if (k0 + 32 < Kdim) {
      stage16(bsrc + (k0 + 32), ldst[buf ^ 1]); // async: next B tile
      an = load_frag(A, Kdim, m0, k0 + 32);     // prefetch next A fragment
      WAIT_ASYNC(1);                            // current tile's copies done
    } else {
      WAIT_ASYNC(0);
    }
    __syncthreads();
    #pragma unroll
    for (int u = 0; u < 4; ++u) {
      v16bf bfr = load_frag_lds(&bsh[buf][16 * u][0]);
      acc[u] = wmma_bf16(a, bfr, acc[u]);
    }
    __syncthreads();                            // tile fully consumed
    a = an;
    buf ^= 1;
  }

  #pragma unroll
  for (int u = 0; u < 4; ++u) {
    int col = n0 + 16 * u + ln;
    float be = (EPI >= 1) ? bias[col] : 0.f;
    #pragma unroll
    for (int r = 0; r < 8; ++r) {
      size_t row = (size_t)m0 + r + 8 * hf;
      float v = acc[u][r] + be;
      if (EPI == 2) v = 0.5f * v * (1.0f + erff(v * 0.70710678118f));
      size_t idx = row * (size_t)N + col;
      if (EPI == 3) outf[idx] = v + resid[idx];
      else          outb[idx] = (bf16)v;
    }
  }
}

// ---------------------------------------------------------------------------
// Row-wise LayerNorm over D=768. grid = BM, block = 256.
// ---------------------------------------------------------------------------
__global__ void ln_kernel(const float* __restrict__ y, const float* __restrict__ g,
                          const float* __restrict__ be, float* __restrict__ outf,
                          bf16* __restrict__ outb) {
  __shared__ float rs1[256], rs2[256];
  const float* p = y + (size_t)blockIdx.x * C_D;
  float a = 0.f, b2 = 0.f;
  for (int c = threadIdx.x; c < C_D; c += 256) { float v = p[c]; a += v; b2 += v * v; }
  rs1[threadIdx.x] = a; rs2[threadIdx.x] = b2;
  __syncthreads();
  for (int o = 128; o > 0; o >>= 1) {
    if ((int)threadIdx.x < o) {
      rs1[threadIdx.x] += rs1[threadIdx.x + o];
      rs2[threadIdx.x] += rs2[threadIdx.x + o];
    }
    __syncthreads();
  }
  float mean = rs1[0] * (1.0f / C_D);
  float var  = rs2[0] * (1.0f / C_D) - mean * mean;
  float rstd = rsqrtf(var + 1e-5f);
  for (int c = threadIdx.x; c < C_D; c += 256) {
    float v = (p[c] - mean) * rstd * g[c] + be[c];
    size_t idx = (size_t)blockIdx.x * C_D + c;
    outf[idx] = v;
    if (outb) outb[idx] = (bf16)v;
  }
}

// ---------------------------------------------------------------------------
// Host launcher
// ---------------------------------------------------------------------------
extern "C" void kernel_launch(void* const* d_in, const int* in_sizes, int n_in,
                              void* d_out, int out_size, void* d_ws, size_t ws_size,
                              hipStream_t stream) {
  const float*   x    = (const float*)d_in[0];
  const uint8_t* mask = (const uint8_t*)d_in[1];
  const float *Pq = (const float*)d_in[2],  *Vq = (const float*)d_in[3],  *bq = (const float*)d_in[4];
  const float *Pk = (const float*)d_in[5],  *Vk = (const float*)d_in[6],  *bk = (const float*)d_in[7];
  const float *Pv = (const float*)d_in[8],  *Vv = (const float*)d_in[9],  *bv = (const float*)d_in[10];
  const float *Uo = (const float*)d_in[11], *Vo = (const float*)d_in[12], *bo = (const float*)d_in[13];
  const float *U1 = (const float*)d_in[14], *V1 = (const float*)d_in[15], *b1 = (const float*)d_in[16];
  const float *U2 = (const float*)d_in[17], *V2 = (const float*)d_in[18], *b2 = (const float*)d_in[19];
  const float *g1 = (const float*)d_in[20], *be1 = (const float*)d_in[21];
  const float *g2 = (const float*)d_in[22], *be2 = (const float*)d_in[23];

  char* wsp = (char*)d_ws;
  size_t off = 0;
  auto alloc = [&](size_t bytes) -> void* {
    void* p = wsp + off;
    off += (bytes + 255) & ~(size_t)255;
    return p;
  };
  bf16* xb    = (bf16*)alloc((size_t)C_BM * C_D * 2);
  bf16* Wt    = (bf16*)alloc((size_t)3 * C_H * C_DH * C_D * 2);
  bf16* Qb    = (bf16*)alloc((size_t)C_B * C_H * C_M * C_DH * 2);
  bf16* Kb    = (bf16*)alloc((size_t)C_B * C_H * C_M * C_DH * 2);
  bf16* Vtb   = (bf16*)alloc((size_t)C_B * C_H * C_DH * C_M * 2);
  bf16* attnb = (bf16*)alloc((size_t)C_BM * C_D * 2);
  bf16* Uo_t  = (bf16*)alloc((size_t)256 * C_D * 2);
  bf16* Vo_t  = (bf16*)alloc((size_t)C_D * 256 * 2);
  bf16* U1_t  = (bf16*)alloc((size_t)256 * C_D * 2);
  bf16* V1_t  = (bf16*)alloc((size_t)C_DFF * 256 * 2);
  bf16* U2_t  = (bf16*)alloc((size_t)256 * C_DFF * 2);
  bf16* V2_t  = (bf16*)alloc((size_t)C_D * 256 * 2);
  bf16* tmid  = (bf16*)alloc((size_t)C_BM * 256 * 2);
  float* yf   = (float*)alloc((size_t)C_BM * C_D * 4);
  float* x1f  = (float*)alloc((size_t)C_BM * C_D * 4);
  bf16* x1b   = (bf16*)alloc((size_t)C_BM * C_D * 2);
  bf16* hb    = (bf16*)alloc((size_t)C_BM * C_DFF * 2);
  (void)ws_size; (void)in_sizes; (void)n_in; (void)out_size;

  // --- prep: conversions + weight folding/transposition
  cvt_bf16_kernel<<<(C_BM * C_D + 255) / 256, 256, 0, stream>>>(x, xb, (size_t)C_BM * C_D);
  prep_wqkv_kernel<<<(3 * C_H * C_DH * C_D + 255) / 256, 256, 0, stream>>>(Pq, Vq, Pk, Vk, Pv, Vv, Wt);
  transpose_bf16_kernel<<<(C_D * 256 + 255) / 256, 256, 0, stream>>>(Uo, Uo_t, C_D, 256);
  transpose_bf16_kernel<<<(256 * C_D + 255) / 256, 256, 0, stream>>>(Vo, Vo_t, 256, C_D);
  transpose_bf16_kernel<<<(C_D * 256 + 255) / 256, 256, 0, stream>>>(U1, U1_t, C_D, 256);
  transpose_bf16_kernel<<<(256 * C_DFF + 255) / 256, 256, 0, stream>>>(V1, V1_t, 256, C_DFF);
  transpose_bf16_kernel<<<(C_DFF * 256 + 255) / 256, 256, 0, stream>>>(U2, U2_t, C_DFF, 256);
  transpose_bf16_kernel<<<(256 * C_D + 255) / 256, 256, 0, stream>>>(V2, V2_t, 256, C_D);

  // --- QKV (fused low-rank weights), then flash attention
  qkv_kernel<<<dim3(C_BM / 128, 3 * C_H), 256, 0, stream>>>(xb, Wt, bq, bk, bv, Qb, Kb, Vtb);
  attn_kernel<<<dim3(C_M / 128, C_B * C_H), 256, 0, stream>>>(Qb, Kb, Vtb, mask, attnb);

  // --- output projection (low rank) + residual + LN1
  gemm_kernel<0><<<dim3(C_BM / 128, 256 / 64), 256, 0, stream>>>(attnb, Uo_t, nullptr, nullptr, nullptr, tmid, 256, C_D);
  gemm_kernel<3><<<dim3(C_BM / 128, C_D / 64), 256, 0, stream>>>(tmid, Vo_t, bo, x, yf, nullptr, C_D, 256);
  ln_kernel<<<C_BM, 256, 0, stream>>>(yf, g1, be1, x1f, x1b);

  // --- low-rank FFN + residual + LN2
  gemm_kernel<0><<<dim3(C_BM / 128, 256 / 64), 256, 0, stream>>>(x1b, U1_t, nullptr, nullptr, nullptr, tmid, 256, C_D);
  gemm_kernel<2><<<dim3(C_BM / 128, C_DFF / 64), 256, 0, stream>>>(tmid, V1_t, b1, nullptr, nullptr, hb, C_DFF, 256);
  gemm_kernel<0><<<dim3(C_BM / 128, 256 / 64), 256, 0, stream>>>(hb, U2_t, nullptr, nullptr, nullptr, tmid, 256, C_DFF);
  gemm_kernel<3><<<dim3(C_BM / 128, C_D / 64), 256, 0, stream>>>(tmid, V2_t, b2, x1f, yf, nullptr, C_D, 256);
  ln_kernel<<<C_BM, 256, 0, stream>>>(yf, g2, be2, (float*)d_out, nullptr);
}
